// SAGPool_57509612093940
// MI455X (gfx1250) — matrix operand order repaired and checked by reading, not compiled, not run on hardware
//
#include <hip/hip_runtime.h>
#include <hip/hip_bf16.h>
#include <math.h>

// ---- problem constants (match reference) ----
#define NB      100          // graphs
#define NPG     1000         // nodes per graph
#define NNODES  (NB * NPG)   // 100000
#define DFEAT   256
#define NEDGE   (NNODES * 32) // 3,200,000
#define KKEEP   800          // ceil(0.8 * NPG)
#define NPOOL   (NB * KKEEP) // 80000

typedef __attribute__((ext_vector_type(16))) _Float16 v16h;
typedef __attribute__((ext_vector_type(8)))  float    v8f;

// ---------------------------------------------------------------------------
// 0) init: deg = 1 (self loop), node_new = -1
// ---------------------------------------------------------------------------
__global__ void sag_init(float* __restrict__ deg, int* __restrict__ node_new) {
    int i = blockIdx.x * blockDim.x + threadIdx.x;
    if (i < NNODES) { deg[i] = 1.0f; node_new[i] = -1; }
}

// ---------------------------------------------------------------------------
// 1) h = x @ W via WMMA f32_16x16x32_f16.
//    One wave handles 16 nodes. A = 16x32 f16 tile of x, B = 32x16 with W
//    replicated across all 16 columns, C accumulates f32 over 8 K-steps.
//    Per CDNA5 C-layout, lane 0 holds M=0..7 (VGPR0..7), lane 16 holds M=8..15.
// ---------------------------------------------------------------------------
__global__ void sag_score_wmma(const float* __restrict__ x,
                               const float* __restrict__ W,
                               float* __restrict__ h) {
    const int wave = blockIdx.x * (blockDim.x >> 5) + (threadIdx.x >> 5);
    const int lane = threadIdx.x & 31;
    const int base = wave * 16;                 // first node of this wave's tile
    if (base >= NNODES) return;                 // wave-uniform
    const int m  = base + (lane & 15);          // node this lane loads A-rows for
    const int hi = lane >> 4;                   // lane half selects K sub-ranges
    const float* __restrict__ xrow = x + (size_t)m * DFEAT;

    v8f c = {};
    #pragma unroll
    for (int kk = 0; kk < DFEAT; kk += 32) {
        v16h a, b;
        #pragma unroll
        for (int e = 0; e < 16; ++e) {
            // A 16x32 f16 layout: lanes 0-15: K = {0..7, 16..23};
            //                     lanes 16-31: K = {8..15, 24..31}
            const int ka = ((e < 8) ? e : e + 8) + hi * 8;
            a[e] = (_Float16)xrow[kk + ka];
            // B 32x16 f16 layout: same K order as A (lanes 0-15: K = {0..7, 16..23}; lanes 16-31: K = {8..15, 24..31});
            // replicate W across all 16 N-columns.
            const int kb = ((e < 8) ? e : e + 8) + hi * 8;
            b[e] = (_Float16)W[kk + kb];
        }
        c = __builtin_amdgcn_wmma_f32_16x16x32_f16(
                /*neg_a=*/false, a, /*neg_b=*/false, b,
                /*c_mod=*/(short)0, c, /*reuse_a=*/false, /*reuse_b=*/false);
    }
    if (lane == 0) {
        #pragma unroll
        for (int r = 0; r < 8; ++r) h[base + r] = c[r];
    } else if (lane == 16) {
        #pragma unroll
        for (int r = 0; r < 8; ++r) h[base + 8 + r] = c[r];
    }
}

// ---------------------------------------------------------------------------
// 2) degree scatter-add over edge targets
// ---------------------------------------------------------------------------
__global__ void sag_degree(const int* __restrict__ edge, float* __restrict__ deg) {
    int e = blockIdx.x * blockDim.x + threadIdx.x;
    if (e < NEDGE) atomicAdd(&deg[edge[NEDGE + e]], 1.0f);
}

// ---------------------------------------------------------------------------
// 3) self-loop contribution + bias:  score[i] = h[i]/deg[i] + b
// ---------------------------------------------------------------------------
__global__ void sag_score_self(const float* __restrict__ h,
                               const float* __restrict__ deg,
                               const float* __restrict__ bias,
                               float* __restrict__ score) {
    int i = blockIdx.x * blockDim.x + threadIdx.x;
    if (i < NNODES) {
        float d = __frsqrt_rn(deg[i]);
        score[i] = d * d * h[i] + bias[0];
    }
}

// ---------------------------------------------------------------------------
// 4) edge aggregation:  score[col] += dinv[row]*dinv[col]*h[row]
// ---------------------------------------------------------------------------
__global__ void sag_aggregate(const int* __restrict__ edge,
                              const float* __restrict__ h,
                              const float* __restrict__ deg,
                              float* __restrict__ score) {
    int e = blockIdx.x * blockDim.x + threadIdx.x;
    if (e < NEDGE) {
        int r = edge[e];
        int c = edge[NEDGE + e];
        float nrm = __frsqrt_rn(deg[r]) * __frsqrt_rn(deg[c]);
        atomicAdd(&score[c], nrm * h[r]);
    }
}

// ---------------------------------------------------------------------------
// 5) per-graph top-K via LDS bitonic sort (1024 slots, descending),
//    emits perm (int + float out), new_batch, node_new relabel table.
// ---------------------------------------------------------------------------
__global__ void sag_topk(const float* __restrict__ score,
                         int* __restrict__ perm_int,
                         int* __restrict__ node_new,
                         float* __restrict__ out_batch,
                         float* __restrict__ out_perm) {
    __shared__ float s_val[1024];
    __shared__ int   s_idx[1024];
    const int g   = blockIdx.x;     // graph
    const int tid = threadIdx.x;    // 256 threads

    for (int i = tid; i < 1024; i += 256) {
        s_val[i] = (i < NPG) ? score[g * NPG + i] : -INFINITY;
        s_idx[i] = i;
    }
    __syncthreads();

    // bitonic sort, descending by value
    for (int k = 2; k <= 1024; k <<= 1) {
        for (int j = k >> 1; j > 0; j >>= 1) {
            for (int i = tid; i < 1024; i += 256) {
                int ixj = i ^ j;
                if (ixj > i) {
                    bool desc = ((i & k) == 0);
                    float a = s_val[i], b = s_val[ixj];
                    bool swap = desc ? (a < b) : (a > b);
                    if (swap) {
                        s_val[i] = b; s_val[ixj] = a;
                        int t = s_idx[i]; s_idx[i] = s_idx[ixj]; s_idx[ixj] = t;
                    }
                }
            }
            __syncthreads();
        }
    }

    for (int j = tid; j < KKEEP; j += 256) {
        int node = g * NPG + s_idx[j];
        int nid  = g * KKEEP + j;
        perm_int[nid]  = node;
        node_new[node] = nid;
        out_perm[nid]  = (float)node;
        out_batch[nid] = (float)g;   // graphs are block-contiguous
    }
}

// ---------------------------------------------------------------------------
// 6) x_pooled[j,:] = x[perm[j],:] * tanh(score[perm[j]])
// ---------------------------------------------------------------------------
__global__ void sag_pool(const float* __restrict__ x,
                         const float* __restrict__ score,
                         const int* __restrict__ perm_int,
                         float* __restrict__ out_x) {
    const int j    = blockIdx.x;          // pooled row, NPOOL blocks
    const int node = perm_int[j];
    const float gate = tanhf(score[node]);
    const float* __restrict__ src = x + (size_t)node * DFEAT;
    float* __restrict__ dst = out_x + (size_t)j * DFEAT;
    for (int f = threadIdx.x; f < DFEAT; f += blockDim.x)
        dst[f] = src[f] * gate;
}

// ---------------------------------------------------------------------------
// 7) edge relabel + mask (outputs written as float values)
// ---------------------------------------------------------------------------
__global__ void sag_filter_adj(const int* __restrict__ edge,
                               const int* __restrict__ node_new,
                               float* __restrict__ out_edge,
                               float* __restrict__ out_mask) {
    int e = blockIdx.x * blockDim.x + threadIdx.x;
    if (e < NEDGE) {
        int a = node_new[edge[e]];
        int c = node_new[edge[NEDGE + e]];
        bool keep = (a >= 0) && (c >= 0);
        out_edge[e]         = (float)(keep ? a : -1);
        out_edge[NEDGE + e] = (float)(keep ? c : -1);
        out_mask[e]         = keep ? 1.0f : 0.0f;
    }
}

// ---------------------------------------------------------------------------
extern "C" void kernel_launch(void* const* d_in, const int* in_sizes, int n_in,
                              void* d_out, int out_size, void* d_ws, size_t ws_size,
                              hipStream_t stream) {
    const float* x    = (const float*)d_in[0];   // [N, D]
    const int*   edge = (const int*)  d_in[1];   // [2, E] flat
    // d_in[2] = batch (unused: graphs block-contiguous)
    const float* W    = (const float*)d_in[3];   // [D]
    const float* bias = (const float*)d_in[4];   // [1]

    // workspace layout
    char* ws = (char*)d_ws;
    float* h        = (float*)ws;                         ws += (size_t)NNODES * 4;
    float* deg      = (float*)ws;                         ws += (size_t)NNODES * 4;
    float* score    = (float*)ws;                         ws += (size_t)NNODES * 4;
    int*   node_new = (int*)ws;                           ws += (size_t)NNODES * 4;
    int*   perm_int = (int*)ws;                           ws += (size_t)NPOOL * 4;

    // output layout (reference tuple order, flattened as floats)
    float* out_x     = (float*)d_out;                      // NPOOL*DFEAT
    float* out_edge  = out_x    + (size_t)NPOOL * DFEAT;   // 2*NEDGE
    float* out_mask  = out_edge + (size_t)2 * NEDGE;       // NEDGE
    float* out_batch = out_mask + (size_t)NEDGE;           // NPOOL
    float* out_perm  = out_batch + (size_t)NPOOL;          // NPOOL

    const int tN = 256;
    // 0) init
    sag_init<<<(NNODES + tN - 1) / tN, tN, 0, stream>>>(deg, node_new);
    // 1) projection h = x@W via WMMA; 2 waves/block, 16 nodes/wave
    sag_score_wmma<<<NNODES / 32, 64, 0, stream>>>(x, W, h);
    // 2) degree
    sag_degree<<<(NEDGE + tN - 1) / tN, tN, 0, stream>>>(edge, deg);
    // 3) self-loop + bias
    sag_score_self<<<(NNODES + tN - 1) / tN, tN, 0, stream>>>(h, deg, bias, score);
    // 4) edge aggregation
    sag_aggregate<<<(NEDGE + tN - 1) / tN, tN, 0, stream>>>(edge, h, deg, score);
    // 5) per-graph top-K
    sag_topk<<<NB, 256, 0, stream>>>(score, perm_int, node_new, out_batch, out_perm);
    // 6) pooled features
    sag_pool<<<NPOOL, 256, 0, stream>>>(x, score, perm_int, out_x);
    // 7) edge filter / relabel
    sag_filter_adj<<<(NEDGE + tN - 1) / tN, tN, 0, stream>>>(edge, node_new,
                                                             out_edge, out_mask);
    (void)in_sizes; (void)n_in; (void)out_size; (void)ws_size;
}
